// DGI_43069932044741
// MI455X (gfx1250) — matrix operand order repaired and checked by reading, not compiled, not run on hardware
//
#include <hip/hip_runtime.h>
#include <hip/hip_bf16.h>
#include <math.h>

#define D 128
#define LDST 136   // padded LDS row stride (bf16 elems): 68 dwords -> conflict-free ds_load_b128

typedef __attribute__((ext_vector_type(16))) __bf16 v16bf;
typedef __attribute__((ext_vector_type(8)))  float  v8f;
typedef __attribute__((ext_vector_type(4)))  int    v4i;

#define AS1 __attribute__((address_space(1)))
#define AS3 __attribute__((address_space(3)))

#if defined(__AMDGCN__) && __has_builtin(__builtin_amdgcn_global_load_async_to_lds_b128) && __has_builtin(__builtin_amdgcn_s_wait_asynccnt)
#define USE_ASYNC_LDS 1
#else
#define USE_ASYNC_LDS 0
#endif

// ---------------- CSR construction ----------------
__global__ void k_zero_int(int* p, int n) {
    int i = blockIdx.x * blockDim.x + threadIdx.x;
    if (i < n) p[i] = 0;
}

__global__ void k_count(const long long* __restrict__ ei, int* __restrict__ cnt, int E) {
    int e = blockIdx.x * blockDim.x + threadIdx.x;
    if (e < E) atomicAdd(&cnt[(int)ei[(long long)E + e]], 1);   // col = target
}

__global__ void k_dinv(const int* __restrict__ cnt, float* __restrict__ dinv, int N) {
    int i = blockIdx.x * blockDim.x + threadIdx.x;
    if (i < N) dinv[i] = rsqrtf((float)cnt[i] + 1.0f);          // +1 self loop
}

// single-block exclusive scan: rowptr[0..N], cursor copy
__global__ void k_scan(const int* __restrict__ cnt, int* __restrict__ rowptr,
                       int* __restrict__ cursor, int N) {
    __shared__ int sums[256];
    const int t = threadIdx.x;
    const int chunk = (N + 255) / 256;
    const int b = t * chunk;
    const int e = min(b + chunk, N);
    int s = 0;
    for (int i = b; i < e; ++i) s += cnt[i];
    sums[t] = s;
    __syncthreads();
    for (int off = 1; off < 256; off <<= 1) {
        int v = (t >= off) ? sums[t - off] : 0;
        __syncthreads();
        sums[t] += v;
        __syncthreads();
    }
    int offset = (t == 0) ? 0 : sums[t - 1];
    for (int i = b; i < e; ++i) {
        rowptr[i] = offset;
        cursor[i] = offset;
        offset += cnt[i];
    }
    if (t == 255) rowptr[N] = offset;
}

__global__ void k_fill(const long long* __restrict__ ei, const float* __restrict__ dinv,
                       int* __restrict__ cursor, int* __restrict__ srcIdx,
                       float* __restrict__ enorm, int E) {
    int e = blockIdx.x * blockDim.x + threadIdx.x;
    if (e >= E) return;
    int r = (int)ei[e];
    int c = (int)ei[(long long)E + e];
    int k = atomicAdd(&cursor[c], 1);
    srcIdx[k] = r;
    enorm[k] = dinv[r] * dinv[c];
}

// ---------------- W (f32, [K,Nc]) -> W^T bf16 ([Nc,K]) ----------------
__global__ void k_w_to_bf16_t(const float* __restrict__ W, __bf16* __restrict__ WT) {
    int i = blockIdx.x * blockDim.x + threadIdx.x;
    if (i < D * D) {
        int k = i >> 7, n = i & 127;
        WT[n * D + k] = (__bf16)W[k * D + n];
    }
}

// ---------------- GEMM: XW = bf16(X) @ W  via v_wmma_f32_16x16x32_bf16 ----------------
static __device__ __forceinline__ v16bf ld_b16x16(const __bf16* p) {
    union { int4 q[2]; v16bf v; } u;
    u.q[0] = *(const int4*)p;
    u.q[1] = *(const int4*)(p + 8);
    return u.v;
}

__global__ void __launch_bounds__(256) k_gemm_wmma(const float* __restrict__ X,
                                                   const __bf16* __restrict__ WT,
                                                   float* __restrict__ XW, int N) {
    __shared__ __bf16 sWT[D * LDST];           // 34.8 KB, padded rows
    const int tid = threadIdx.x;

    // stage W^T (128x128 bf16) into LDS once per block
#if USE_ASYNC_LDS
#pragma unroll
    for (int i = 0; i < 8; ++i) {
        int j = tid + 256 * i;                 // 2048 int4 chunks total
        int r = j >> 4, cc = j & 15;
        __builtin_amdgcn_global_load_async_to_lds_b128(
            (AS1 v4i*)(WT + (size_t)r * D + cc * 8),
            (AS3 v4i*)(sWT + (size_t)r * LDST + cc * 8), 0, 0);
    }
    __builtin_amdgcn_s_wait_asynccnt(0);
    __syncthreads();
#else
#pragma unroll
    for (int i = 0; i < 8; ++i) {
        int j = tid + 256 * i;
        int r = j >> 4, cc = j & 15;
        *(int4*)(sWT + (size_t)r * LDST + cc * 8) =
            *(const int4*)(WT + (size_t)r * D + cc * 8);
    }
    __syncthreads();
#endif

    const int lane = tid & 31;
    const int wave = tid >> 5;
    const int half = lane >> 4;
    const int l15  = lane & 15;
    const int row0 = blockIdx.x * 128 + wave * 16;

    int arow = row0 + l15;
    if (arow > N - 1) arow = N - 1;            // clamp; stores are guarded
    const float* xr = X + (size_t)arow * D;

    v8f zero = {0.f, 0.f, 0.f, 0.f, 0.f, 0.f, 0.f, 0.f};
    v8f acc[8];
#pragma unroll
    for (int t = 0; t < 8; ++t) acc[t] = zero;

#pragma unroll
    for (int ks = 0; ks < 4; ++ks) {
        const int k0 = ks * 32;
        v16bf a;
        const float* p0 = xr + k0 + half * 16;
        const float* p1 = xr + k0 + half * 16 + 8;
#pragma unroll
        for (int i = 0; i < 8; ++i) a[i] = (__bf16)p0[i];
#pragma unroll
        for (int i = 0; i < 8; ++i) a[8 + i] = (__bf16)p1[i];

#pragma unroll
        for (int t = 0; t < 8; ++t) {
            const int n = t * 16 + l15;        // B: lane holds column n, 16 contiguous K
            v16bf b = ld_b16x16(sWT + (size_t)n * LDST + k0 + half * 16);
            acc[t] = __builtin_amdgcn_wmma_f32_16x16x32_bf16(
                false, a, false, b, (short)0, acc[t], false, false);
        }
    }

#pragma unroll
    for (int t = 0; t < 8; ++t) {
        const int n = t * 16 + l15;
#pragma unroll
        for (int e = 0; e < 8; ++e) {
            const int rr = row0 + half * 8 + e;
            if (rr < N) XW[(size_t)rr * D + n] = acc[t][e];
        }
    }
}

// ---------------- fused gather aggregation + bias + PReLU ----------------
// out[c] = prelu( dinv[c]^2 * xw[c] + sum_{e:(r->c)} dinv[r]*dinv[c]*xw[r] + b , a )
__global__ void __launch_bounds__(256) k_gather(const int* __restrict__ rowptr,
        const int* __restrict__ srcIdx, const float* __restrict__ enorm,
        const float* __restrict__ dinv, const float* __restrict__ xw,
        const float* __restrict__ bias, const float* __restrict__ a_ptr,
        float* __restrict__ out, int N) {
    int gid = blockIdx.x * blockDim.x + threadIdx.x;
    int node = gid >> 5, lane = gid & 31;
    if (node >= N) return;

    float di = dinv[node];
    float self = di * di;
    const float4 sv = *(const float4*)(xw + (size_t)node * D + lane * 4);
    float4 acc = {self * sv.x, self * sv.y, self * sv.z, self * sv.w};

    const int beg = rowptr[node], end = rowptr[node + 1];
    for (int base = beg; base < end; base += 32) {
        int k = base + lane;
        int rsrc = 0; float nm = 0.f;
        if (k < end) { rsrc = srcIdx[k]; nm = enorm[k]; }
        int cnt = end - base; if (cnt > 32) cnt = 32;
        for (int j = 0; j < cnt; ++j) {
            int   rr = __shfl(rsrc, j, 32);
            float nj = __shfl(nm,  j, 32);
            const float4 v = *(const float4*)(xw + (size_t)rr * D + lane * 4);
            acc.x += nj * v.x; acc.y += nj * v.y;
            acc.z += nj * v.z; acc.w += nj * v.w;
        }
    }

    const float a = *a_ptr;
    float4 bb = *(const float4*)(bias + lane * 4);
    acc.x += bb.x; acc.y += bb.y; acc.z += bb.z; acc.w += bb.w;
    acc.x = (acc.x >= 0.f) ? acc.x : a * acc.x;
    acc.y = (acc.y >= 0.f) ? acc.y : a * acc.y;
    acc.z = (acc.z >= 0.f) ? acc.z : a * acc.z;
    acc.w = (acc.w >= 0.f) ? acc.w : a * acc.w;
    *(float4*)(out + (size_t)node * D + lane * 4) = acc;
}

// ---------------- readout + bilinear ----------------
__global__ void k_zero128(float* p) { p[threadIdx.x] = 0.0f; }

__global__ void k_col_accum(const float* __restrict__ h, float* __restrict__ csum, int N) {
    int col = threadIdx.x;                 // 128 threads
    int r0 = blockIdx.x * 128;
    int rend = r0 + 128; if (rend > N) rend = N;
    float s = 0.f;
    for (int r = r0; r < rend; ++r) s += h[(size_t)r * D + col];
    atomicAdd(&csum[col], s);
}

__global__ void k_make_wc(const float* __restrict__ csum, const float* __restrict__ Wb,
                          float* __restrict__ wc, int N) {
    __shared__ float cs[D];
    int j = threadIdx.x;                   // 128 threads
    cs[j] = 1.f / (1.f + expf(-csum[j] / (float)N));
    __syncthreads();
    float acc = 0.f;
#pragma unroll 4
    for (int k = 0; k < D; ++k) acc += Wb[j * D + k] * cs[k];
    wc[j] = acc;
}

__global__ void k_scores(const float* __restrict__ h1, const float* __restrict__ h2,
                         const float* __restrict__ wc, const float* __restrict__ bb,
                         float* __restrict__ out, int N) {
    int gid = blockIdx.x * blockDim.x + threadIdx.x;
    int r = gid >> 5, lane = gid & 31;
    if (r >= 2 * N) return;
    const float* h = (r < N) ? (h1 + (size_t)r * D) : (h2 + (size_t)(r - N) * D);
    float s = 0.f;
#pragma unroll
    for (int i = 0; i < 4; ++i) s += h[lane + 32 * i] * wc[lane + 32 * i];
#pragma unroll
    for (int off = 16; off > 0; off >>= 1) s += __shfl_xor(s, off, 32);
    if (lane == 0) out[r] = s + bb[0];
}

// ---------------- host ----------------
static inline void* carve(char*& p, size_t bytes) {
    void* r = (void*)p;
    p += (bytes + 255) & ~(size_t)255;
    return r;
}

extern "C" void kernel_launch(void* const* d_in, const int* in_sizes, int n_in,
                              void* d_out, int out_size, void* d_ws, size_t ws_size,
                              hipStream_t stream) {
    const float*     seq1 = (const float*)d_in[0];
    const float*     seq2 = (const float*)d_in[1];
    const long long* ei   = (const long long*)d_in[2];
    const float* W1 = (const float*)d_in[3];
    const float* b1 = (const float*)d_in[4];
    const float* W2 = (const float*)d_in[5];
    const float* b2 = (const float*)d_in[6];
    const float* W3 = (const float*)d_in[7];
    const float* b3 = (const float*)d_in[8];
    const float* a_prelu = (const float*)d_in[9];
    const float* Wb      = (const float*)d_in[10];
    const float* b_bilin = (const float*)d_in[11];

    const int N = in_sizes[0] / D;
    const int E = in_sizes[2] / 2;

    char* p = (char*)d_ws;
    float*  dinv   = (float*)carve(p, (size_t)N * 4);
    int*    cnt    = (int*)carve(p, (size_t)N * 4);
    int*    rowptr = (int*)carve(p, (size_t)(N + 1) * 4);
    int*    cursor = (int*)carve(p, (size_t)N * 4);
    int*    srcIdx = (int*)carve(p, (size_t)E * 4);
    float*  enorm  = (float*)carve(p, (size_t)E * 4);
    float*  csum   = (float*)carve(p, 128 * 4);
    float*  wc     = (float*)carve(p, 128 * 4);
    __bf16* WT1    = (__bf16*)carve(p, (size_t)D * D * 2);
    __bf16* WT2    = (__bf16*)carve(p, (size_t)D * D * 2);
    __bf16* WT3    = (__bf16*)carve(p, (size_t)D * D * 2);
    float*  xw     = (float*)carve(p, (size_t)N * D * 4);
    float*  bufA   = (float*)carve(p, (size_t)N * D * 4);
    float*  bufB   = (float*)carve(p, (size_t)N * D * 4);
    float*  h1     = (float*)carve(p, (size_t)N * D * 4);
    float*  h2     = (float*)carve(p, (size_t)N * D * 4);

    const int T = 256;
    const int gN    = (N + T - 1) / T;
    const int gE    = (E + T - 1) / T;
    const int gW    = (D * D + T - 1) / T;
    const int gRowV = (N * 32 + T - 1) / T;     // one wave per node
    const int gGemm = (N + 127) / 128;

    // CSR + normalization (rebuilt every call; no cached state)
    k_zero_int<<<gN, T, 0, stream>>>(cnt, N);
    k_count<<<gE, T, 0, stream>>>(ei, cnt, E);
    k_dinv<<<gN, T, 0, stream>>>(cnt, dinv, N);
    k_scan<<<1, 256, 0, stream>>>(cnt, rowptr, cursor, N);
    k_fill<<<gE, T, 0, stream>>>(ei, dinv, cursor, srcIdx, enorm, E);

    // weights -> bf16 transposed
    k_w_to_bf16_t<<<gW, T, 0, stream>>>(W1, WT1);
    k_w_to_bf16_t<<<gW, T, 0, stream>>>(W2, WT2);
    k_w_to_bf16_t<<<gW, T, 0, stream>>>(W3, WT3);

    // one GCN layer: out = prelu(Aggr(in @ W) + b)   (gather-fused, no atomics)
    auto layer = [&](const float* in, const __bf16* WT, const float* b, float* out) {
        k_gemm_wmma<<<gGemm, T, 0, stream>>>(in, WT, xw, N);
        k_gather<<<gRowV, T, 0, stream>>>(rowptr, srcIdx, enorm, dinv, xw, b,
                                          a_prelu, out, N);
    };

    layer(seq1, WT1, b1, bufA);
    layer(bufA, WT2, b2, bufB);
    layer(bufB, WT3, b3, h1);
    layer(seq2, WT1, b1, bufA);
    layer(bufA, WT2, b2, bufB);
    layer(bufB, WT3, b3, h2);

    // readout c = sigmoid(mean(h1)); wc = W_bilin @ c
    k_zero128<<<1, 128, 0, stream>>>(csum);
    k_col_accum<<<(N + 127) / 128, 128, 0, stream>>>(h1, csum, N);
    k_make_wc<<<1, 128, 0, stream>>>(csum, Wb, wc, N);

    // scores
    k_scores<<<(2 * N * 32 + T - 1) / T, T, 0, stream>>>(h1, h2, wc, b_bilin,
                                                         (float*)d_out, N);
}
